// EncGraph_62740882260319
// MI455X (gfx1250) — matrix operand order repaired and checked by reading, not compiled, not run on hardware
//
#include <hip/hip_runtime.h>

// ---------------------------------------------------------------------------
// Problem constants (from reference)
// ---------------------------------------------------------------------------
#define TN   8192   // time steps
#define NN   64     // nodes per graph
#define FF   128    // input features
#define HH   128    // encoder hidden
#define OO   128    // gcn output

typedef unsigned int u32;
typedef unsigned short u16;
typedef __attribute__((ext_vector_type(2)))  u32    u32x2;
typedef __attribute__((ext_vector_type(4)))  u32    u32x4;
typedef __attribute__((ext_vector_type(4)))  float  f32x4;
typedef __attribute__((ext_vector_type(8)))  float  v8f;
typedef __attribute__((ext_vector_type(16))) __bf16 v16bf;

// 16 bf16 = 8 VGPRs: A/B operand of v_wmma_f32_16x16x32_bf16
union FragAB {
    v16bf  v;
    u32x4  q[2];
    __bf16 b[16];
};

// 4 bf16 packed for 8-byte LDS stores
union Pack4 {
    __bf16 b[4];
    u32x2  w;
};

// float -> bf16 (native convert) -> raw 16 bits
static __device__ __forceinline__ u16 bf_bits(float f) {
    union { __bf16 b; u16 s; } u;
    u.b = (__bf16)f;
    return u.s;
}

// A fragment (16x32, MxK) from row-major bf16 matrix, leading dim ld.
// ISA layout: lanes 0-15 row M=lane, VGPR0..3 = K 0..7, VGPR4..7 = K 16..23;
// lanes 16-31 same rows, K +8.  => two contiguous 16B chunks per lane.
static __device__ __forceinline__ FragAB load_a_bf(const u16* M, int ld,
                                                   int row0, int k0, int lane) {
    FragAB r;
    const int m = lane & 15, hi = lane >> 4;
    const u16* p = M + (size_t)(row0 + m) * ld + k0 + hi * 8;
    r.q[0] = *(const u32x4*)(p);        // K = k0 + hi*8 + 0..7
    r.q[1] = *(const u32x4*)(p + 16);   // K = k0 + 16 + hi*8 + 0..7
    return r;
}

// B fragment (32x16, KxN) from *transposed* row-major storage BT[N][K], ld.
// ISA layout: lanes 0-15 col N=lane, VGPR0..3 = K 0..7, VGPR4..7 = K 16..23; lanes 16-31 K +8.
// => two contiguous 16B chunks per lane from row (col0+n) of BT.
static __device__ __forceinline__ FragAB load_b_t(const u16* BT, int ld,
                                                  int col0, int k0, int lane) {
    FragAB r;
    const int n = lane & 15, hi = lane >> 4;
    const u16* p = BT + (size_t)(col0 + n) * ld + k0 + hi * 8;
    r.q[0] = *(const u32x4*)(p);        // K = k0 + hi*8 + 0..7
    r.q[1] = *(const u32x4*)(p + 16);   // K = k0 + 16 + hi*8 + 0..7
    return r;
}

static __device__ __forceinline__ v8f wmma_bf16(const FragAB& a, const FragAB& b, v8f c) {
    return __builtin_amdgcn_wmma_f32_16x16x32_bf16(
        /*neg_a=*/false, a.v, /*neg_b=*/false, b.v,
        /*c_mod=*/(short)0, c, /*reuse_a=*/false, /*reuse_b=*/false);
}

// ---------------------------------------------------------------------------
// Prep kernel (1 block): normalized adjacency (bf16) + transposed bf16 weights
// norm = D_in^-1/2 (A + I) D_out^-1/2  with A[dst,src] += 1 per edge
// ---------------------------------------------------------------------------
__global__ __launch_bounds__(256) void prep_kernel(
    const float* __restrict__ W_enc, const float* __restrict__ W_gcn,
    const int* __restrict__ src, const int* __restrict__ dst, int E,
    __bf16* __restrict__ normbf, __bf16* __restrict__ WencT, __bf16* __restrict__ WgcnT)
{
    __shared__ float A[NN * NN];
    __shared__ float rin[NN], rout[NN];
    const int tid = threadIdx.x;

    for (int i = tid; i < NN * NN; i += 256) A[i] = 0.0f;
    __syncthreads();
    for (int e = tid; e < E; e += 256)
        atomicAdd(&A[dst[e] * NN + src[e]], 1.0f);
    if (tid < NN)
        atomicAdd(&A[tid * NN + tid], 1.0f);            // self loop
    __syncthreads();
    if (tid < NN) {                                     // in-degree (row sum)
        float s = 0.0f;
        for (int j = 0; j < NN; ++j) s += A[tid * NN + j];
        rin[tid] = rsqrtf(s);
    } else if (tid < 2 * NN) {                          // out-degree (col sum)
        const int j = tid - NN;
        float s = 0.0f;
        for (int i = 0; i < NN; ++i) s += A[i * NN + j];
        rout[j] = rsqrtf(s);
    }
    __syncthreads();
    for (int i = tid; i < NN * NN; i += 256)
        normbf[i] = (__bf16)(A[i] * rin[i >> 6] * rout[i & 63]);

    // Transposed bf16 weights: WT[out][in] = W[in][out]
    for (int i = tid; i < FF * HH; i += 256) {
        const int o = i >> 7, in = i & 127;
        WencT[i] = (__bf16)W_enc[in * HH + o];
        WgcnT[i] = (__bf16)W_gcn[in * OO + o];
    }
}

// ---------------------------------------------------------------------------
// Fused kernel: one block per time step.
//   stage 0: X_t (f32, global) --coalesced b128--> bf16 LDS (converted once)
//   H   = X_t @ W_enc + b_enc        (A = Xbf LDS,   B = WencT,  D -> LDS H^T)
//   AGG = norm @ H                   (A = norm bf16, B^T = H^T in LDS)
//   OUT = AGG @ W_gcn + b_gcn        (A = AGG LDS,   B = WgcnT,  D -> global)
// ---------------------------------------------------------------------------
#define LDX 136   // Xbf  leading dim (128 feat + pad),  rows 16B-aligned (272B)
#define LDH 72    // H^T  leading dim (64 nodes + pad),  rows 16B-aligned (144B)
#define LDA 136   // AGG  leading dim (128 hidden + pad), rows 16B-aligned (272B)

__global__ __launch_bounds__(256) void fused_kernel(
    const float* __restrict__ x,
    const float* __restrict__ b_enc, const float* __restrict__ b_gcn,
    const u16* __restrict__ normbf, const u16* __restrict__ WencT,
    const u16* __restrict__ WgcnT, float* __restrict__ out)
{
    __shared__ __align__(16) u16 Xbf[NN * LDX];    // X_t : [node][feat] bf16
    __shared__ __align__(16) u16 HbfT[HH * LDH];   // H^T : [hidden][node]
    __shared__ __align__(16) u16 AggBf[NN * LDA];  // AGG : [node][hidden]

    const int t    = blockIdx.x;
    const int tid  = threadIdx.x;
    const int wave = tid >> 5;
    const int lane = tid & 31;
    const int mlan = lane & 15;
    const int hi   = lane >> 4;
    const int row0 = (wave >> 1) * 16;     // node row tile
    const int nh   = (wave & 1) * 64;      // column half base

    // ---- Stage 0: coalesced X load + single bf16 conversion into LDS ----
    {
        const f32x4* Xv = (const f32x4*)(x + (size_t)t * (NN * FF));
#pragma unroll
        for (int it = 0; it < 8; ++it) {
            const int idx = tid + it * 256;        // float4 index, 2048 total
            const f32x4 v = Xv[idx];
            const int row = idx >> 5;              // 32 float4 per row
            const int col = (idx & 31) * 4;
            Pack4 pk;
            pk.b[0] = (__bf16)v.x; pk.b[1] = (__bf16)v.y;
            pk.b[2] = (__bf16)v.z; pk.b[3] = (__bf16)v.w;
            *(u32x2*)&Xbf[(size_t)row * LDX + col] = pk.w;
        }
    }
    __syncthreads();

    v8f acc[4];

    // ---------------- Stage A: H = X @ Wenc + b_enc -> HbfT ----------------
#pragma unroll
    for (int i = 0; i < 4; ++i) {
        const float bv = b_enc[nh + i * 16 + mlan];
#pragma unroll
        for (int v = 0; v < 8; ++v) acc[i][v] = bv;
    }
#pragma unroll
    for (int k0 = 0; k0 < FF; k0 += 32) {
        const FragAB a = load_a_bf(Xbf, LDX, row0, k0, lane);
#pragma unroll
        for (int i = 0; i < 4; ++i) {
            const FragAB b = load_b_t(WencT, FF, nh + i * 16, k0, lane);
            acc[i] = wmma_bf16(a, b, acc[i]);
        }
    }
    // D tile (m,n): store transposed -> HbfT[n][m], 8 bf16 contiguous per lane
#pragma unroll
    for (int i = 0; i < 4; ++i) {
        FragAB pk;
#pragma unroll
        for (int v = 0; v < 8; ++v) pk.b[v] = (__bf16)acc[i][v];
        *(u32x4*)&HbfT[(size_t)(nh + i * 16 + mlan) * LDH + row0 + hi * 8] = pk.q[0];
    }
    __syncthreads();

    // ---------------- Stage B: AGG = norm @ H -> AggBf ----------------
#pragma unroll
    for (int i = 0; i < 4; ++i)
#pragma unroll
        for (int v = 0; v < 8; ++v) acc[i][v] = 0.0f;
#pragma unroll
    for (int k0 = 0; k0 < NN; k0 += 32) {
        const FragAB a = load_a_bf(normbf, NN, row0, k0, lane);
#pragma unroll
        for (int i = 0; i < 4; ++i) {
            const FragAB b = load_b_t(HbfT, LDH, nh + i * 16, k0, lane);
            acc[i] = wmma_bf16(a, b, acc[i]);
        }
    }
#pragma unroll
    for (int i = 0; i < 4; ++i)
#pragma unroll
        for (int v = 0; v < 8; ++v)
            AggBf[(size_t)(row0 + hi * 8 + v) * LDA + nh + i * 16 + mlan] =
                bf_bits(acc[i][v]);
    __syncthreads();

    // ---------------- Stage C: OUT = AGG @ Wgcn + b_gcn -> global ----------------
#pragma unroll
    for (int i = 0; i < 4; ++i) {
        const float bv = b_gcn[nh + i * 16 + mlan];
#pragma unroll
        for (int v = 0; v < 8; ++v) acc[i][v] = bv;
    }
#pragma unroll
    for (int k0 = 0; k0 < HH; k0 += 32) {
        const FragAB a = load_a_bf(AggBf, LDA, row0, k0, lane);
#pragma unroll
        for (int i = 0; i < 4; ++i) {
            const FragAB b = load_b_t(WgcnT, HH, nh + i * 16, k0, lane);
            acc[i] = wmma_bf16(a, b, acc[i]);
        }
    }
    float* obase = out + ((size_t)t * NN + row0 + hi * 8) * OO + nh;
#pragma unroll
    for (int i = 0; i < 4; ++i)
#pragma unroll
        for (int v = 0; v < 8; ++v)
            obase[(size_t)v * OO + i * 16 + mlan] = acc[i][v];
}

// ---------------------------------------------------------------------------
extern "C" void kernel_launch(void* const* d_in, const int* in_sizes, int n_in,
                              void* d_out, int out_size, void* d_ws, size_t ws_size,
                              hipStream_t stream) {
    const float* x     = (const float*)d_in[0];
    const float* W_enc = (const float*)d_in[1];
    const float* b_enc = (const float*)d_in[2];
    const float* W_gcn = (const float*)d_in[3];
    const float* b_gcn = (const float*)d_in[4];
    const int*   src   = (const int*)d_in[5];
    const int*   dst   = (const int*)d_in[6];
    const int    E     = in_sizes[5];
    float* out = (float*)d_out;

    // Workspace: norm bf16 (8KB) | WencT bf16 (32KB) | WgcnT bf16 (32KB)
    __bf16* normbf = (__bf16*)d_ws;
    __bf16* WencT  = normbf + NN * NN;
    __bf16* WgcnT  = WencT + FF * HH;

    prep_kernel<<<1, 256, 0, stream>>>(W_enc, W_gcn, src, dst, E,
                                       normbf, WencT, WgcnT);
    fused_kernel<<<TN, 256, 0, stream>>>(x, b_enc, b_gcn,
                                         (const u16*)normbf, (const u16*)WencT,
                                         (const u16*)WgcnT, out);
}
